// RNN_90400471646408
// MI455X (gfx1250) — matrix operand order repaired and checked by reading, not compiled, not run on hardware
//
#include <hip/hip_runtime.h>
#include <hip/hip_bf16.h>

// ---------------------------------------------------------------------------
// 2-layer tanh RNN, T=1024, B=64, H=512, fp32 in/out, bf16 WMMA compute.
//
// MI455X strategy, round 4:
//  * 4 batch-groups x {layer-0 producer, layer-1 consumer} WG pairs, joined by
//    a 4-deep h0 ring in L2 with acquire/release counters (no grid barrier).
//  * K-loop: two 16-chunk half-loops, dynamic induction variable (prevents the
//    round-2 hoist+scratch-spill), #pragma unroll 4 for a deep clause window.
//  * Activation: hardware V_TANH_F32 via inline asm (ISA lists TANH as a
//    TRANS32 op; no clang builtin). v_nop embedded per the trans-op hazard
//    rule. Replaces the ~20-op OCML tanh chain on the critical path.
//  * Per-tile bias sums hoisted out of the t-loop.
//  * Weights fused (ih|hh) + converted once to bf16 in d_ws (2 MB, L2-hot).
//  * Hidden state double-buffered in LDS (32 KB of 320 KB/WGP).
// ---------------------------------------------------------------------------

typedef __attribute__((ext_vector_type(16))) __bf16 v16bf;
typedef __attribute__((ext_vector_type(8)))  __bf16 v8bf;
typedef __attribute__((ext_vector_type(8)))  float  v8f;

#define T_STEPS 1024
#define BATCH   64
#define HID     512
#define LAYERS  2
#define KDIM    1024            // HID (first half) + HID (recurrent half)
#define NGROUP  4               // batch groups, 16 rows each
#define ROWS_PER_WG 16
#define THREADS 256             // 8 waves (wave32)
#define NTILES_PER_WAVE 4       // 512 cols / 16 / 8 waves
#define RING    4               // h0 handoff ring depth (power of two)

// ---- hardware tanh (V_TANH_F32, TRANS32 op) --------------------------------
// The compiler cannot model inline-asm latency, so per ISA 7.4 (a trans-op result
// needs one intervening op before it is read) a v_nop is embedded.
__device__ __forceinline__ float hw_tanhf(float x) {
  float r;
  asm("v_tanh_f32 %0, %1\n\tv_nop" : "=v"(r) : "v"(x));
  return r;
}

// ---- WMMA fragment loaders -------------------------------------------------
// A (16x32 bf16): lane holds row lane&15; K = k0..k0+15 with
// k0 = c + 16*(lane>=16).
__device__ __forceinline__ v16bf a_from_f32(const float* __restrict__ p, int k0) {
  const float4 f0 = *(const float4*)(p + k0);
  const float4 f1 = *(const float4*)(p + k0 + 4);
  const float4 f2 = *(const float4*)(p + k0 + 8);
  const float4 f3 = *(const float4*)(p + k0 + 12);
  v16bf a;
  a[0]  = (__bf16)f0.x; a[1]  = (__bf16)f0.y; a[2]  = (__bf16)f0.z; a[3]  = (__bf16)f0.w;
  a[4]  = (__bf16)f1.x; a[5]  = (__bf16)f1.y; a[6]  = (__bf16)f1.z; a[7]  = (__bf16)f1.w;
  a[8]  = (__bf16)f2.x; a[9]  = (__bf16)f2.y; a[10] = (__bf16)f2.z; a[11] = (__bf16)f2.w;
  a[12] = (__bf16)f3.x; a[13] = (__bf16)f3.y; a[14] = (__bf16)f3.z; a[15] = (__bf16)f3.w;
  return a;
}

__device__ __forceinline__ v16bf a_from_bf16(const __bf16* __restrict__ p, int k0) {
  const v8bf lo = *(const v8bf*)(p + k0);
  const v8bf hi = *(const v8bf*)(p + k0 + 8);
  return __builtin_shufflevector(lo, hi, 0, 1, 2, 3, 4, 5, 6, 7,
                                         8, 9, 10, 11, 12, 13, 14, 15);
}

// ---- producer/consumer handoff --------------------------------------------
__device__ __forceinline__ void wait_flag_ge(const int* p, int target) {
  if (threadIdx.x == 0) {
    while (__hip_atomic_load(p, __ATOMIC_ACQUIRE, __HIP_MEMORY_SCOPE_AGENT) < target)
      __builtin_amdgcn_s_sleep(1);
  }
  __syncthreads();
}

__device__ __forceinline__ void signal_flag(int* p, int v) {
  __syncthreads();  // all waves of this WG finished their stores/reads
  if (threadIdx.x == 0)
    __hip_atomic_store(p, v, __ATOMIC_RELEASE, __HIP_MEMORY_SCOPE_AGENT);
}

// ---- half-GEMM over K=[0,512): 16 chunks of 32, dynamic loop --------------
// XF32 selects fp32 (x_t) vs bf16 source for the A fragment.
template <bool XF32>
__device__ __forceinline__ void gemm_half(
    const float*  __restrict__ xrow,    // per-lane fp32 row (XF32 only)
    const __bf16* __restrict__ arow,    // per-lane bf16 row (!XF32 only)
    const __bf16* __restrict__ w0,      // lane's weight row base for tile 0
    v8f acc[NTILES_PER_WAVE],
    int k0a)
{
  // Tiles are 16 cols apart => weight row bases are 16*KDIM elements apart.
#pragma unroll 4
  for (int kc = 0; kc < HID / 32; ++kc) {
    const int c = kc * 32;              // dynamic: loads cannot be hoisted
    v16bf a;
    if (XF32) a = a_from_f32(xrow, c + k0a);
    else      a = a_from_bf16(arow, c + k0a);
#pragma unroll
    for (int j = 0; j < NTILES_PER_WAVE; ++j) {
      const v16bf b = *(const v16bf*)(w0 + (size_t)j * 16 * KDIM + c);
      acc[j] = __builtin_amdgcn_wmma_f32_16x16x32_bf16(
          false, a, false, b, (short)0, acc[j], false, false);
    }
  }
}

// ---- one layer, one timestep: D[16x512] = tanh(A[16x1024] @ W^T + b) -------
template <bool XF32, bool HAS_RING, bool HAS_SEQ>
__device__ __forceinline__ void layer_gemm_step(
    const float*  __restrict__ xrow,    // per-lane fp32 row (XF32)
    const __bf16* __restrict__ a0row,   // per-lane bf16 row, K in [0,512)
    const __bf16* __restrict__ a1row,   // per-lane bf16 row, K in [512,1024)
    const __bf16* __restrict__ Wl,      // [512][1024] bf16 (fused ih|hh)
    const float*  __restrict__ bsum,    // per-tile bias sums (registers)
    __bf16*       __restrict__ hO,      // LDS [16][512]
    __bf16*       __restrict__ ringO,   // ring slot [16][512] (HAS_RING)
    float*        __restrict__ seqO,    // out + t*B*H (HAS_SEQ)
    int grow_base)
{
  const int lane = threadIdx.x & 31;
  const int wave = threadIdx.x >> 5;
  const int k0a  = (lane >> 4) << 4;    // +16 K offset, upper half-wave (A)
  const int kb0  = (lane >> 4) << 4;    // +16 K offset, upper half-wave (B)
  const int nl   = lane & 15;

  const v8f vz = {0.f, 0.f, 0.f, 0.f, 0.f, 0.f, 0.f, 0.f};
  v8f acc[NTILES_PER_WAVE] = {vz, vz, vz, vz};

  // Lane's weight row base for this wave's first tile.
  const __bf16* w0 = Wl + (size_t)(wave * NTILES_PER_WAVE * 16 + nl) * KDIM + kb0;

  gemm_half<XF32>(xrow, a0row, w0, acc, k0a);           // K in [0,512)
  gemm_half<false>(nullptr, a1row, w0 + HID, acc, k0a); // K in [512,1024)

  // D layout: lane<16 -> M=v, lane>=16 -> M=8+v; N = lane&15 within the tile.
  const int rbase = (lane >> 4) << 3;
#pragma unroll
  for (int j = 0; j < NTILES_PER_WAVE; ++j) {
    const int nc = (wave * NTILES_PER_WAVE + j) * 16 + nl;
#pragma unroll
    for (int v = 0; v < 8; ++v) {
      const float f = hw_tanhf(acc[j][v] + bsum[j]);
      const int r = rbase + v;
      hO[r * HID + nc] = (__bf16)f;
      if (HAS_RING) ringO[r * HID + nc] = (__bf16)f;
      if (HAS_SEQ)  seqO[((size_t)(grow_base + r)) * HID + nc] = f;
    }
  }
}

// ---- init: fuse W_ih|W_hh -> bf16 Wcat[l][n][k] and zero handoff flags -----
__global__ __launch_bounds__(256) void rnn_convert_weights(
    const float* __restrict__ W_ih, const float* __restrict__ W_hh,
    __bf16* __restrict__ Wbf, int* __restrict__ flags)
{
  if (blockIdx.x == 0 && threadIdx.x < NGROUP * 32)
    flags[threadIdx.x] = 0;
  const int idx4 = (blockIdx.x * 256 + threadIdx.x) * 4;
  if (idx4 >= LAYERS * HID * KDIM) return;
  const int l   = idx4 / (HID * KDIM);
  const int rem = idx4 % (HID * KDIM);
  const int n   = rem / KDIM;
  const int k   = rem % KDIM;
  const float* src = (k < HID) ? (W_ih + ((size_t)l * HID + n) * HID + k)
                               : (W_hh + ((size_t)l * HID + n) * HID + (k - HID));
  const float4 v = *(const float4*)src;
  __bf16* d = Wbf + idx4;
  d[0] = (__bf16)v.x; d[1] = (__bf16)v.y; d[2] = (__bf16)v.z; d[3] = (__bf16)v.w;
}

// ---- persistent pipelined RNN ----------------------------------------------
__global__ __launch_bounds__(THREADS, 1) void rnn_pipeline(
    const float* __restrict__ x,       // [T][B][H] fp32
    const float* __restrict__ b_ih,    // [L][H]
    const float* __restrict__ b_hh,    // [L][H]
    const __bf16* __restrict__ Wbf,    // [L][512][1024] bf16
    float* __restrict__ out,           // [T][B][H] ++ [L][B][H]
    __bf16* __restrict__ ring,         // [NGROUP][RING][16][512] bf16
    int* __restrict__ flags)           // [NGROUP][32]: produced@0, consumed@16
{
  extern __shared__ char smem[];
  __bf16* hbuf = (__bf16*)smem;        // [2 parity][16][512] bf16 = 32 KB

  const int tid   = threadIdx.x;
  const int group = blockIdx.x >> 1;
  const int role  = blockIdx.x & 1;    // 0 = layer-0 producer, 1 = layer-1 consumer
  const int lane  = tid & 31;
  const int wave  = tid >> 5;
  const int rA    = lane & 15;
  const int grow_base = group * ROWS_PER_WG;

  int* produced = flags + group * 32;
  int* consumed = flags + group * 32 + 16;
  __bf16* gring = ring + (size_t)group * RING * ROWS_PER_WG * HID;

  for (int i = tid; i < 2 * ROWS_PER_WG * HID; i += THREADS)
    hbuf[i] = (__bf16)0.0f;            // h0 = 0
  __syncthreads();

  // Per-tile bias sums for this role's layer: t-invariant, keep in registers.
  float bsum[NTILES_PER_WAVE];
#pragma unroll
  for (int j = 0; j < NTILES_PER_WAVE; ++j) {
    const int nc = (wave * NTILES_PER_WAVE + j) * 16 + (lane & 15);
    bsum[j] = b_ih[role * HID + nc] + b_hh[role * HID + nc];
  }

  float* hx = out + (size_t)T_STEPS * BATCH * HID;

  if (role == 0) {
    // ---- layer-0 producer: h0(t) = tanh(x_t @ Wih0^T + h0(t-1) @ Whh0^T + b)
    for (int t = 0; t < T_STEPS; ++t) {
      const int cur = t & 1, prev = cur ^ 1;
      wait_flag_ge(consumed, t - (RING - 1));          // ring slot free?
      const float* xrow = x + ((size_t)t * BATCH + grow_base + rA) * HID;
      if (t + 1 < T_STEPS)
        __builtin_prefetch(x + ((size_t)(t + 1) * BATCH + grow_base + rA) * HID, 0, 1);
      layer_gemm_step<true, true, false>(
          xrow, nullptr,
          hbuf + (prev * ROWS_PER_WG + rA) * HID,
          Wbf, bsum,
          hbuf + cur * ROWS_PER_WG * HID,
          gring + (size_t)(t & (RING - 1)) * ROWS_PER_WG * HID,
          nullptr, grow_base);
      signal_flag(produced, t + 1);
    }
    for (int i = tid; i < ROWS_PER_WG * HID; i += THREADS) {
      const int r = i / HID, c = i % HID;              // final parity = 1
      hx[((size_t)grow_base + r) * HID + c] =
          (float)hbuf[(1 * ROWS_PER_WG + r) * HID + c];
    }
  } else {
    // ---- layer-1 consumer: h1(t) = tanh(h0(t) @ Wih1^T + h1(t-1) @ Whh1^T + b)
    for (int t = 0; t < T_STEPS; ++t) {
      const int cur = t & 1, prev = cur ^ 1;
      wait_flag_ge(produced, t + 1);                   // h0(t) ready?
      const __bf16* h0row =
          gring + (size_t)(t & (RING - 1)) * ROWS_PER_WG * HID + rA * HID;
      layer_gemm_step<false, false, true>(
          nullptr, h0row,
          hbuf + (prev * ROWS_PER_WG + rA) * HID,
          Wbf + (size_t)HID * KDIM,
          bsum,
          hbuf + cur * ROWS_PER_WG * HID,
          nullptr,
          out + (size_t)t * BATCH * HID, grow_base);
      signal_flag(consumed, t + 1);
    }
    for (int i = tid; i < ROWS_PER_WG * HID; i += THREADS) {
      const int r = i / HID, c = i % HID;              // final parity = 1
      hx[((size_t)BATCH + grow_base + r) * HID + c] =
          (float)hbuf[(1 * ROWS_PER_WG + r) * HID + c];
    }
  }
}

extern "C" void kernel_launch(void* const* d_in, const int* in_sizes, int n_in,
                              void* d_out, int out_size, void* d_ws, size_t ws_size,
                              hipStream_t stream) {
  (void)in_sizes; (void)n_in; (void)out_size; (void)ws_size;
  const float* x    = (const float*)d_in[0];
  const float* W_ih = (const float*)d_in[1];
  const float* W_hh = (const float*)d_in[2];
  const float* b_ih = (const float*)d_in[3];
  const float* b_hh = (const float*)d_in[4];
  float* out = (float*)d_out;

  // d_ws layout: [0, 2MB) fused bf16 weights | [2MB, +256KB) h0 ring | flags
  const size_t W_BYTES    = (size_t)LAYERS * HID * KDIM * sizeof(__bf16);
  const size_t RING_BYTES = (size_t)NGROUP * RING * ROWS_PER_WG * HID * sizeof(__bf16);
  __bf16* Wbf  = (__bf16*)d_ws;
  __bf16* ring = (__bf16*)((char*)d_ws + W_BYTES);
  int*    flags = (int*)((char*)d_ws + W_BYTES + RING_BYTES);

  // Per-call weight fuse/convert + flag reset (deterministic, graph-safe).
  rnn_convert_weights<<<1024, 256, 0, stream>>>(W_ih, W_hh, Wbf, flags);

  const int shmem = 2 * ROWS_PER_WG * HID * (int)sizeof(__bf16);  // 32 KB
  rnn_pipeline<<<2 * NGROUP, THREADS, shmem, stream>>>(
      x, b_ih, b_hh, Wbf, out, ring, flags);
}
